// CrossWindowAttention_2954937500123
// MI455X (gfx1250) — compile-verified, matrix operand order checked by reading, not run on hardware
//
#include <hip/hip_runtime.h>
#include <hip/hip_bf16.h>

typedef __attribute__((ext_vector_type(16))) _Float16 v16h;
typedef __attribute__((ext_vector_type(8)))  _Float16 v8h;
typedef __attribute__((ext_vector_type(4)))  _Float16 v4h;
typedef __attribute__((ext_vector_type(8)))  float    v8f;
typedef __attribute__((ext_vector_type(4)))  float    v4f;

#define HEADS_ 3
#define N_     64
#define DIM_   96
#define HD_    32
#define BWIN_  8192
#define NWIN_  2048
#define SCALE_ 0.17677669529663687f   /* 32^-0.5 */

/* ---- workspace layout (bytes) ---- */
#define WS_WQ1T 0            /* f16 [192][96]  = 36864 B */
#define WS_WQ2T 36864        /* f16 [96][96]   = 18432 B */
#define WS_WPT  55296        /* f16 [96][96]   = 18432 B */
#define WS_RPB  73728        /* f32 [3][64][64]= 49152 B */

/* ---- dynamic LDS layout (bytes); all 16B aligned ----
   o_lds aliases x_lds: x dead after phase B, o first written in phase F. */
#define L_X   0              /* f16 [64][104]  (later reused as o_lds)      */
#define L_Y   13312          /* f16 [64][104]                               */
#define L_Q   26624          /* f16 [64][104]                               */
#define L_K   39936          /* f16 [64][104]                               */
#define L_VT  53248          /* f16 [96][72] (row = out-col h*32+d, col=key)*/
#define L_P   67072          /* f16 [3][64][72]                             */
#define L_A   94720          /* f32 [64][64] per-head logits scratch        */
#define LDS_BYTES 111104     /* -> 2 workgroups per WGP (320KB)             */

/* Load a 16x32 f16 A/B fragment (B passed as B^T rows) from row-major storage.
   CDNA5 16-bit A layout: lane<16 -> K 0..7 / 16..23, lane>=16 -> K 8..15 / 24..31. */
__device__ __forceinline__ v16h load_frag(const _Float16* tile, int stride) {
  const int lane = threadIdx.x & 31;
  const int r  = lane & 15;
  const int kc = (lane >> 4) << 3;             /* 0 or 8 */
  const _Float16* p = tile + r * stride + kc;
  v8h lo = *(const v8h*)(p);
  v8h hi = *(const v8h*)(p + 16);
  v16h o;
#pragma unroll
  for (int i = 0; i < 8; ++i) { o[i] = lo[i]; o[i + 8] = hi[i]; }
  return o;
}

__device__ __forceinline__ v8f wmma_f16(v16h a, v16h b, v8f c) {
  return __builtin_amdgcn_wmma_f32_16x16x32_f16(false, a, false, b, (short)0, c,
                                                false, false);
}

/* ---------- prep: transpose+cvt weights to f16, build rpb table ---------- */
__global__ void xwin_prep_kernel(const float* __restrict__ w1,
                                 const float* __restrict__ w2,
                                 const float* __restrict__ wp,
                                 const float* __restrict__ bt,
                                 _Float16* __restrict__ wq1t,
                                 _Float16* __restrict__ wq2t,
                                 _Float16* __restrict__ wpt,
                                 float* __restrict__ rpb) {
  const int tid = blockIdx.x * blockDim.x + threadIdx.x;
  const int stride = gridDim.x * blockDim.x;
  for (int i = tid; i < 192 * 96; i += stride) {          /* wq1t[n][k] = w1[k][n] */
    int n = i / 96, k = i % 96;
    wq1t[i] = (_Float16)w1[k * 192 + n];
  }
  for (int i = tid; i < 96 * 96; i += stride) {
    int n = i / 96, k = i % 96;
    wq2t[i] = (_Float16)w2[k * 96 + n];
    wpt[i]  = (_Float16)wp[k * 96 + n];
  }
  for (int i = tid; i < HEADS_ * N_ * N_; i += stride) {  /* rpb[h][i][j] */
    int h = i >> 12, r = (i >> 6) & 63, c = i & 63;
    int ih = r >> 3, iw = r & 7, jh = c >> 3, jw = c & 7;
    int idx = (ih - jh + 7) * 15 + (iw - jw + 7);
    rpb[i] = bt[idx * HEADS_ + h];
  }
}

/* ---------- main: one block (8 wave32) per window ---------- */
__global__ __launch_bounds__(256)
void xwin_attn_kernel(const float* __restrict__ x, const float* __restrict__ y,
                      const float* __restrict__ mask,
                      const float* __restrict__ b1, const float* __restrict__ b2,
                      const float* __restrict__ bp,
                      const _Float16* __restrict__ wq1t,
                      const _Float16* __restrict__ wq2t,
                      const _Float16* __restrict__ wpt,
                      const float* __restrict__ rpb,
                      float* __restrict__ out) {
  extern __shared__ char smem[];
  _Float16* x_lds = (_Float16*)(smem + L_X);
  _Float16* y_lds = (_Float16*)(smem + L_Y);
  _Float16* q_lds = (_Float16*)(smem + L_Q);
  _Float16* k_lds = (_Float16*)(smem + L_K);
  _Float16* v_t   = (_Float16*)(smem + L_VT);
  _Float16* p_lds = (_Float16*)(smem + L_P);
  float*    a_lds = (float*)   (smem + L_A);
  _Float16* o_lds = (_Float16*)(smem + L_X);   /* alias over dead x */

  const int b     = blockIdx.x;
  const int tid   = threadIdx.x;
  const int wave  = tid >> 5;
  const int lane  = tid & 31;
  const int cl    = lane & 15;
  const int rbase = (lane >> 4) << 3;

  /* --- Phase A: stage x,y -> LDS f16 (row stride 104 halves), packed b64 stores --- */
  {
    const v4f* xb = (const v4f*)(x + (size_t)b * (N_ * DIM_));
    const v4f* yb = (const v4f*)(y + (size_t)b * (N_ * DIM_));
    for (int i = tid; i < N_ * 24; i += 256) {
      int r = i / 24, c = (i % 24) * 4;
      v4f vx = xb[i], vy = yb[i];
      v4h hx, hy;
#pragma unroll
      for (int j = 0; j < 4; ++j) { hx[j] = (_Float16)vx[j]; hy[j] = (_Float16)vy[j]; }
      *(v4h*)(x_lds + r * 104 + c) = hx;
      *(v4h*)(y_lds + r * 104 + c) = hy;
    }
  }
  __syncthreads();

  /* --- Phase B: kv = x @ W1 + b1 ; cols 0..95 -> K, 96..191 -> V (stored transposed) --- */
#pragma unroll
  for (int j = 0; j < 6; ++j) {
    int t = wave * 6 + j;
    int m = t / 12, n = t % 12;
    float bias = b1[n * 16 + cl];
    v8f acc;
#pragma unroll
    for (int i = 0; i < 8; ++i) acc[i] = bias;
#pragma unroll
    for (int ks = 0; ks < 3; ++ks) {
      v16h a  = load_frag(x_lds + (m * 16) * 104 + ks * 32, 104);
      v16h bm = load_frag(wq1t + (n * 16) * 96 + ks * 32, 96);
      acc = wmma_f16(a, bm, acc);
    }
    int col = n * 16 + cl;
    int r0  = m * 16 + rbase;
    if (n < 6) {
#pragma unroll
      for (int v = 0; v < 8; ++v) k_lds[(r0 + v) * 104 + col] = (_Float16)acc[v];
    } else {
      int d = col - 96;
#pragma unroll
      for (int v = 0; v < 8; ++v) v_t[d * 72 + (r0 + v)] = (_Float16)acc[v];
    }
  }

  /* --- Phase C: q = (y @ W2 + b2) * SCALE --- */
#pragma unroll
  for (int j = 0; j < 3; ++j) {
    int t = wave * 3 + j;
    int m = t / 6, n = t % 6;
    float bias = b2[n * 16 + cl];
    v8f acc;
#pragma unroll
    for (int i = 0; i < 8; ++i) acc[i] = bias;
#pragma unroll
    for (int ks = 0; ks < 3; ++ks) {
      v16h a  = load_frag(y_lds + (m * 16) * 104 + ks * 32, 104);
      v16h bm = load_frag(wq2t + (n * 16) * 96 + ks * 32, 96);
      acc = wmma_f16(a, bm, acc);
    }
    int col = n * 16 + cl;
    int r0  = m * 16 + rbase;
#pragma unroll
    for (int v = 0; v < 8; ++v)
      q_lds[(r0 + v) * 104 + col] = (_Float16)(acc[v] * SCALE_);
  }
  __syncthreads();

  /* --- Phase D/E per head: logits = q@k^T + rpb + mask, then row softmax -> P f16 --- */
  {
    const float* maskb = mask + (size_t)(b & (NWIN_ - 1)) * (N_ * N_);
    for (int h = 0; h < HEADS_; ++h) {
      const float* rp = rpb + h * (N_ * N_);
#pragma unroll
      for (int j = 0; j < 2; ++j) {               /* 16 tiles / 8 waves */
        int t = wave * 2 + j;
        int m = t >> 2, n = t & 3;
        int col = n * 16 + cl;
        int r0  = m * 16 + rbase;
        v8f acc;
#pragma unroll
        for (int v = 0; v < 8; ++v)
          acc[v] = rp[(r0 + v) * 64 + col] + maskb[(r0 + v) * 64 + col];
        v16h a  = load_frag(q_lds + (m * 16) * 104 + h * 32, 104);
        v16h bm = load_frag(k_lds + (n * 16) * 104 + h * 32, 104);
        acc = wmma_f16(a, bm, acc);
#pragma unroll
        for (int v = 0; v < 8; ++v) a_lds[(r0 + v) * 64 + col] = acc[v];
      }
      __syncthreads();
      if (tid < N_) {                             /* one thread per row */
        const float* row = a_lds + tid * 64;
        float mx = row[0];
        for (int c = 1; c < 64; ++c) mx = fmaxf(mx, row[c]);
        float s = 0.f;
        for (int c = 0; c < 64; ++c) s += __expf(row[c] - mx);
        float inv = 1.f / s;
        _Float16* pr = p_lds + (h * N_ + tid) * 72;
        for (int c = 0; c < 64; ++c) pr[c] = (_Float16)(__expf(row[c] - mx) * inv);
      }
      __syncthreads();
    }
  }

  /* --- Phase F: out = P @ V  (B-operand from V^T, K = 64 keys -> 2 steps) --- */
#pragma unroll
  for (int j = 0; j < 3; ++j) {
    int t = wave * 3 + j;
    int m = t / 6, n = t % 6;
    int h = n >> 1;
    v8f acc;
#pragma unroll
    for (int i = 0; i < 8; ++i) acc[i] = 0.f;
#pragma unroll
    for (int ks = 0; ks < 2; ++ks) {
      v16h a  = load_frag(p_lds + (h * N_ + m * 16) * 72 + ks * 32, 72);
      v16h bm = load_frag(v_t + (n * 16) * 72 + ks * 32, 72);
      acc = wmma_f16(a, bm, acc);
    }
    int col = n * 16 + cl;
    int r0  = m * 16 + rbase;
#pragma unroll
    for (int v = 0; v < 8; ++v) o_lds[(r0 + v) * 104 + col] = (_Float16)acc[v];
  }
  __syncthreads();

  /* --- Phase G: final = o @ Wproj + bp -> global --- */
  {
    float* ob = out + (size_t)b * (N_ * DIM_);
#pragma unroll
    for (int j = 0; j < 3; ++j) {
      int t = wave * 3 + j;
      int m = t / 6, n = t % 6;
      float bias = bp[n * 16 + cl];
      v8f acc;
#pragma unroll
      for (int i = 0; i < 8; ++i) acc[i] = bias;
#pragma unroll
      for (int ks = 0; ks < 3; ++ks) {
        v16h a  = load_frag(o_lds + (m * 16) * 104 + ks * 32, 104);
        v16h bm = load_frag(wpt + (n * 16) * 96 + ks * 32, 96);
        acc = wmma_f16(a, bm, acc);
      }
      int col = n * 16 + cl;
      int r0  = m * 16 + rbase;
#pragma unroll
      for (int v = 0; v < 8; ++v) ob[(r0 + v) * 96 + col] = acc[v];
    }
  }
}

extern "C" void kernel_launch(void* const* d_in, const int* in_sizes, int n_in,
                              void* d_out, int out_size, void* d_ws, size_t ws_size,
                              hipStream_t stream) {
  const float* x    = (const float*)d_in[0];
  const float* y    = (const float*)d_in[1];
  const float* mask = (const float*)d_in[2];
  const float* w1   = (const float*)d_in[3];
  const float* b1   = (const float*)d_in[4];
  const float* w2   = (const float*)d_in[5];
  const float* b2   = (const float*)d_in[6];
  const float* bt   = (const float*)d_in[7];
  const float* wp   = (const float*)d_in[8];
  const float* bp   = (const float*)d_in[9];
  float* out = (float*)d_out;

  char* ws = (char*)d_ws;
  _Float16* wq1t = (_Float16*)(ws + WS_WQ1T);
  _Float16* wq2t = (_Float16*)(ws + WS_WQ2T);
  _Float16* wpt  = (_Float16*)(ws + WS_WPT);
  float*    rpb  = (float*)(ws + WS_RPB);

  xwin_prep_kernel<<<64, 256, 0, stream>>>(w1, w2, wp, bt, wq1t, wq2t, wpt, rpb);
  xwin_attn_kernel<<<BWIN_, 256, LDS_BYTES, stream>>>(x, y, mask, b1, b2, bp,
                                                      wq1t, wq2t, wpt, rpb, out);
}
